// ALIGN_50414326120804
// MI455X (gfx1250) — matrix operand order repaired and checked by reading, not compiled, not run on hardware
//
#include <hip/hip_runtime.h>

typedef __attribute__((ext_vector_type(16))) _Float16 v16h;
typedef __attribute__((ext_vector_type(8)))  _Float16 v8h;
typedef __attribute__((ext_vector_type(8)))  float    v8f;

#define HH 128
#define WW 128
#define CIN 64
#define KB 18            // K blocks of 32 (K = 576)
#define NPB 32           // pixels per workgroup
#define MT1 4            // 64 / 16 output-channel tiles
#define MT2 14           // 224 / 16 tiles (216 padded)
#define XSTRIDE 104      // lds halo: per-ci stride (3*34 = 102, padded)
#define BSTRIDE 584      // lds im2col: per-pixel K stride (576 padded, 16B units)

__device__ __forceinline__ v8f wmma16(v16h a, v16h b, v8f c) {
  return __builtin_amdgcn_wmma_f32_16x16x32_f16(false, a, false, b, (short)0, c, false, false);
}

// LDS B-fragment load: 16 contiguous halves, guaranteed 16B (not 32B) aligned.
__device__ __forceinline__ v16h load_bfrag(const _Float16* p) {
  v8h lo = *(const v8h*)p;
  v8h hi = *(const v8h*)(p + 8);
  return __builtin_shufflevector(lo, hi, 0,1,2,3,4,5,6,7,8,9,10,11,12,13,14,15);
}

// ---------------------------------------------------------------------------
// Pack conv weights [Cout][64][3][3] (f32) into A-fragment layout (f16):
//   p[((mt*KB + kb)*32 + lane)*16 + j] = W[mt*16 + lane%16][K(kb,lane,j)]
//   K = kb*32 + (lane>=16 ? 16 : 0) + j   (contiguous k order, matches B)
// ---------------------------------------------------------------------------
__global__ __launch_bounds__(256) void pack_weights_kernel(
    const float* __restrict__ w1, const float* __restrict__ w2,
    const float* __restrict__ w3,
    _Float16* __restrict__ p1, _Float16* __restrict__ p2,
    _Float16* __restrict__ p3) {
  const int S1 = MT1 * KB * 512;       // 36864
  const int S2 = MT2 * KB * 512;       // 129024
  const int S3 = MT1 * KB * 512;       // 36864
  int t = blockIdx.x * 256 + threadIdx.x;
  const float* w; _Float16* p; int idx, cout;
  if (t < S1)            { w = w1; p = p1; idx = t;           cout = 64;  }
  else if (t < S1 + S2)  { w = w2; p = p2; idx = t - S1;      cout = 216; }
  else if (t < S1+S2+S3) { w = w3; p = p3; idx = t - S1 - S2; cout = 64;  }
  else return;
  int j    = idx & 15;
  int lane = (idx >> 4) & 31;
  int kb   = (idx >> 9) % KB;
  int mt   = idx / (KB * 512);
  int co   = mt * 16 + (lane & 15);
  int K    = kb * 32 + ((lane >> 4) ? 16 : 0) + j;
  int ci = K / 9, r = K - ci * 9;
  float v = (co < cout) ? w[(co * CIN + ci) * 9 + r] : 0.f;
  p[idx] = (_Float16)v;
}

// ---------------------------------------------------------------------------
// Conv1: cat_fea (f32) -> offset_feat (f16 in workspace). 64->64, 3x3 SAME.
// ---------------------------------------------------------------------------
__global__ __launch_bounds__(256) void conv1_kernel(
    const float* __restrict__ x, const _Float16* __restrict__ wp,
    const float* __restrict__ bias, _Float16* __restrict__ out) {
  __shared__ _Float16 lx[CIN * XSTRIDE];
  __shared__ _Float16 lb[NPB * BSTRIDE];
  int blk = blockIdx.x;
  int wb = blk & 3, h = (blk >> 2) & (HH - 1), b = blk >> 9;
  int w0 = wb * NPB;
  // stage zero-padded halo tile (64 ci x 3 rows x 34 cols) as f16
  for (int t = threadIdx.x; t < CIN * 3 * 34; t += 256) {
    int cx = t % 34, tt = t / 34, ry = tt % 3, ci = tt / 3;
    int gy = h + ry - 1, gx = w0 + cx - 1;
    float v = 0.f;
    if ((unsigned)gy < HH && (unsigned)gx < WW)
      v = x[((b * CIN + ci) * HH + gy) * WW + gx];
    lx[ci * XSTRIDE + ry * 34 + cx] = (_Float16)v;
  }
  __syncthreads();
  // im2col expand: lb[n][K] = x[ci][h+ry-1][w0+n+rx-1], K = ci*9 + ry*3 + rx
  for (int t = threadIdx.x; t < NPB * 576; t += 256) {
    int K = t % 576, n = t / 576;
    int ci = K / 9, r = K - ci * 9, ry = r / 3, rx = r - ry * 3;
    lb[n * BSTRIDE + K] = lx[ci * XSTRIDE + ry * 34 + n + rx];
  }
  __syncthreads();
  int wid = threadIdx.x >> 5, lane = threadIdx.x & 31;
  int ln = lane & 15, hf = lane >> 4;
  int mt = wid & 3, n0 = (wid >> 2) * 16;
  v8f acc = {0.f, 0.f, 0.f, 0.f, 0.f, 0.f, 0.f, 0.f};
  const _Float16* lbc = lb + (n0 + ln) * BSTRIDE + hf * 16;
  for (int kb = 0; kb < KB; ++kb) {
    v16h a = *(const v16h*)(wp + ((mt * KB + kb) * 32 + lane) * 16);
    v16h bf = load_bfrag(lbc + kb * 32);
    acc = wmma16(a, bf, acc);
  }
  int mb = mt * 16 + hf * 8;
  int px = w0 + n0 + ln;
#pragma unroll
  for (int i = 0; i < 8; ++i) {
    int co = mb + i;
    out[((b * CIN + co) * HH + h) * WW + px] = (_Float16)(acc[i] + bias[co]);
  }
}

// ---------------------------------------------------------------------------
// Conv2: offset_feat (f16) -> 216ch "off"; epilogue routes dy/dx/sigmoid-mask
// into f32 workspace arrays laid out [b][dg*9+kk][h][w].
// ---------------------------------------------------------------------------
__global__ __launch_bounds__(256) void conv2_kernel(
    const _Float16* __restrict__ xf, const _Float16* __restrict__ wp,
    const float* __restrict__ bias, float* __restrict__ dyA,
    float* __restrict__ dxA, float* __restrict__ mkA) {
  __shared__ _Float16 lx[CIN * XSTRIDE];
  __shared__ _Float16 lb[NPB * BSTRIDE];
  int blk = blockIdx.x;
  int wb = blk & 3, h = (blk >> 2) & (HH - 1), b = blk >> 9;
  int w0 = wb * NPB;
  for (int t = threadIdx.x; t < CIN * 3 * 34; t += 256) {
    int cx = t % 34, tt = t / 34, ry = tt % 3, ci = tt / 3;
    int gy = h + ry - 1, gx = w0 + cx - 1;
    _Float16 v = (_Float16)0.f;
    if ((unsigned)gy < HH && (unsigned)gx < WW)
      v = xf[((b * CIN + ci) * HH + gy) * WW + gx];
    lx[ci * XSTRIDE + ry * 34 + cx] = v;
  }
  __syncthreads();
  for (int t = threadIdx.x; t < NPB * 576; t += 256) {
    int K = t % 576, n = t / 576;
    int ci = K / 9, r = K - ci * 9, ry = r / 3, rx = r - ry * 3;
    lb[n * BSTRIDE + K] = lx[ci * XSTRIDE + ry * 34 + n + rx];
  }
  __syncthreads();
  int wid = threadIdx.x >> 5, lane = threadIdx.x & 31;
  int ln = lane & 15, hf = lane >> 4;
  int n0 = (wid >> 2) * 16;
  int px = w0 + n0 + ln;
  const _Float16* lbc = lb + (n0 + ln) * BSTRIDE + hf * 16;
  // keep all 18 B-fragments in registers; reuse across the 3-4 M tiles
  v16h bfs[KB];
#pragma unroll
  for (int kb = 0; kb < KB; ++kb) bfs[kb] = load_bfrag(lbc + kb * 32);
  for (int mt = (wid & 3); mt < MT2; mt += 4) {
    v8f acc = {0.f, 0.f, 0.f, 0.f, 0.f, 0.f, 0.f, 0.f};
#pragma unroll
    for (int kb = 0; kb < KB; ++kb) {
      v16h a = *(const v16h*)(wp + ((mt * KB + kb) * 32 + lane) * 16);
      acc = wmma16(a, bfs[kb], acc);
    }
    int mb = mt * 16 + hf * 8;
#pragma unroll
    for (int i = 0; i < 8; ++i) {
      int co = mb + i;
      if (co >= 216) continue;
      float val = acc[i] + bias[co];
      if (co < 144) {
        // off channels 0..143 reshape to (DG, KK, 2): ch = dg*18 + kk*2 + s
        int dg = co / 18, rem = co % 18, kk = rem >> 1;
        int base = ((b * 72 + dg * 9 + kk) * HH + h) * WW + px;
        if (rem & 1) dxA[base] = val; else dyA[base] = val;
      } else {
        int base = ((b * 72 + (co - 144)) * HH + h) * WW + px;
        mkA[base] = 1.f / (1.f + __expf(-val));
      }
    }
  }
}

// ---------------------------------------------------------------------------
// Deform: bilinear-sample f_fea at offset positions, apply mask, build the
// 576xN value tile in LDS, then the final GEMM (einsum) via WMMA.
// ---------------------------------------------------------------------------
__global__ __launch_bounds__(256) void deform_kernel(
    const float* __restrict__ f_fea, const float* __restrict__ dyA,
    const float* __restrict__ dxA, const float* __restrict__ mkA,
    const _Float16* __restrict__ wp, const float* __restrict__ bias,
    float* __restrict__ out) {
  __shared__ _Float16 lv[NPB * BSTRIDE];
  int blk = blockIdx.x;
  int wb = blk & 3, h = (blk >> 2) & (HH - 1), b = blk >> 9;
  int w0 = wb * NPB;
  // 2304 tasks = exactly 9 iterations for 256 threads (no divergence)
  for (int t = threadIdx.x; t < NPB * 72; t += 256) {
    int dgk = t % 72, n = t / 72;
    int dg = dgk / 9, kk = dgk - dg * 9;
    int w = w0 + n;
    int obase = ((b * 72 + dgk) * HH + h) * WW + w;
    float dy = dyA[obase], dx = dxA[obase], mk = mkA[obase];
    float py = dy + (float)(h + kk / 3 - 1);
    float pxs = dx + (float)(w + kk % 3 - 1);
    float y0f = floorf(py), x0f = floorf(pxs);
    float ly = py - y0f, lx = pxs - x0f;
    int y0 = (int)y0f, x0 = (int)x0f, y1 = y0 + 1, x1 = x0 + 1;
    float vy0 = ((unsigned)y0 < HH) ? 1.f : 0.f;
    float vy1 = ((unsigned)y1 < HH) ? 1.f : 0.f;
    float vx0 = ((unsigned)x0 < WW) ? 1.f : 0.f;
    float vx1 = ((unsigned)x1 < WW) ? 1.f : 0.f;
    int cy0 = min(max(y0, 0), HH - 1), cy1 = min(max(y1, 0), HH - 1);
    int cx0 = min(max(x0, 0), WW - 1), cx1 = min(max(x1, 0), WW - 1);
    float w00 = (1.f - ly) * (1.f - lx) * vy0 * vx0;
    float w01 = (1.f - ly) * lx * vy0 * vx1;
    float w10 = ly * (1.f - lx) * vy1 * vx0;
    float w11 = ly * lx * vy1 * vx1;
    int i00 = cy0 * WW + cx0, i01 = cy0 * WW + cx1;
    int i10 = cy1 * WW + cx0, i11 = cy1 * WW + cx1;
    const float* fb = f_fea + (size_t)(b * CIN + dg * 8) * (HH * WW);
    _Float16* dst = lv + n * BSTRIDE + dg * 72 + kk;  // K = (dg*8+cc)*9 + kk
#pragma unroll
    for (int cc = 0; cc < 8; ++cc) {
      const float* fc = fb + cc * (HH * WW);
      float s = w00 * fc[i00] + w01 * fc[i01] + w10 * fc[i10] + w11 * fc[i11];
      dst[cc * 9] = (_Float16)(s * mk);
    }
  }
  __syncthreads();
  int wid = threadIdx.x >> 5, lane = threadIdx.x & 31;
  int ln = lane & 15, hf = lane >> 4;
  int mt = wid & 3, n0 = (wid >> 2) * 16;
  v8f acc = {0.f, 0.f, 0.f, 0.f, 0.f, 0.f, 0.f, 0.f};
  const _Float16* lvc = lv + (n0 + ln) * BSTRIDE + hf * 16;
  for (int kb = 0; kb < KB; ++kb) {
    v16h a = *(const v16h*)(wp + ((mt * KB + kb) * 32 + lane) * 16);
    v16h bf = load_bfrag(lvc + kb * 32);
    acc = wmma16(a, bf, acc);
  }
  int mb = mt * 16 + hf * 8;
  int px = w0 + n0 + ln;
#pragma unroll
  for (int i = 0; i < 8; ++i) {
    int co = mb + i;
    out[((b * CIN + co) * HH + h) * WW + px] = acc[i] + bias[co];
  }
}

// ---------------------------------------------------------------------------
extern "C" void kernel_launch(void* const* d_in, const int* in_sizes, int n_in,
                              void* d_out, int out_size, void* d_ws,
                              size_t ws_size, hipStream_t stream) {
  const float* cat_fea = (const float*)d_in[0];
  const float* f_fea   = (const float*)d_in[1];
  const float* w_off2d = (const float*)d_in[2];
  const float* b_off2d = (const float*)d_in[3];
  const float* w_coff  = (const float*)d_in[4];
  const float* b_coff  = (const float*)d_in[5];
  const float* w_dconv = (const float*)d_in[6];
  const float* b_dconv = (const float*)d_in[7];
  float* out = (float*)d_out;

  char* ws = (char*)d_ws;
  const size_t SZ_OFF  = (size_t)8 * 64 * HH * WW * 2;   // offset_feat f16
  const size_t SZ_CH72 = (size_t)8 * 72 * HH * WW * 4;   // dy/dx/mask f32
  _Float16* offF = (_Float16*)ws;
  float* dyA = (float*)(ws + SZ_OFF);
  float* dxA = (float*)(ws + SZ_OFF + SZ_CH72);
  float* mkA = (float*)(ws + SZ_OFF + 2 * SZ_CH72);
  _Float16* p1 = (_Float16*)(ws + SZ_OFF + 3 * SZ_CH72);
  _Float16* p2 = p1 + (size_t)MT1 * KB * 512;
  _Float16* p3 = p2 + (size_t)MT2 * KB * 512;

  int packTot = (MT1 + MT2 + MT1) * KB * 512;  // 202752
  pack_weights_kernel<<<(packTot + 255) / 256, 256, 0, stream>>>(
      w_off2d, w_coff, w_dconv, p1, p2, p3);

  int grid = 8 * HH * (WW / NPB);  // 4096 workgroups
  conv1_kernel<<<grid, 256, 0, stream>>>(cat_fea, p1, b_off2d, offF);
  conv2_kernel<<<grid, 256, 0, stream>>>(offF, p2, b_coff, dyA, dxA, mkA);
  deform_kernel<<<grid, 256, 0, stream>>>(f_fea, dyA, dxA, mkA, p3, b_dconv,
                                          out);
}
